// ContinuousMambaBlock_4449586119089
// MI455X (gfx1250) — matrix operand order repaired and checked by reading, not compiled, not run on hardware
//
#include <hip/hip_runtime.h>
#include <hip/hip_bf16.h>

typedef __attribute__((ext_vector_type(16))) __bf16 v16bf;
typedef __attribute__((ext_vector_type(8)))  float  v8f;

// ---------------------------------------------------------------------------
// bf16 fragment packing (CDNA5 ISA 7.12.2 layouts)
// ---------------------------------------------------------------------------
__device__ __forceinline__ v16bf pack16(float4 x0, float4 x1, float4 x2, float4 x3) {
  v16bf v;
  v[0]=(__bf16)x0.x; v[1]=(__bf16)x0.y; v[2]=(__bf16)x0.z; v[3]=(__bf16)x0.w;
  v[4]=(__bf16)x1.x; v[5]=(__bf16)x1.y; v[6]=(__bf16)x1.z; v[7]=(__bf16)x1.w;
  v[8]=(__bf16)x2.x; v[9]=(__bf16)x2.y; v[10]=(__bf16)x2.z; v[11]=(__bf16)x2.w;
  v[12]=(__bf16)x3.x; v[13]=(__bf16)x3.y; v[14]=(__bf16)x3.z; v[15]=(__bf16)x3.w;
  return v;
}

// A-matrix 16x32 (16-bit): lane-half h holds K = {k0+8h..k0+8h+7, k0+16+8h..+7}
__device__ __forceinline__ v16bf load_a_frag(const float* __restrict__ row, int k0, int half) {
  const float4* p1 = (const float4*)(row + k0 + half * 8);
  const float4* p2 = (const float4*)(row + k0 + 16 + half * 8);
  return pack16(p1[0], p1[1], p2[0], p2[1]);
}

// B-matrix 32x16 (16-bit), column n per lane: lane-half h holds K = {k0+8h..k0+8h+7, k0+16+8h..+7}
__device__ __forceinline__ v16bf load_b_frag(const float* __restrict__ wrow, int k0, int half) {
  const float4* p1 = (const float4*)(wrow + k0 + half * 8);
  const float4* p2 = (const float4*)(wrow + k0 + 16 + half * 8);
  return pack16(p1[0], p1[1], p2[0], p2[1]);
}

// C/D layout: VGPR rr -> row row0 + rr + 8*half, col fixed per lane
template <int MODE>
__device__ __forceinline__ void store_tile(v8f acc, float* __restrict__ Cout,
                                           const float* __restrict__ bias,
                                           const float* __restrict__ aux,
                                           int row0, int col, int half, int N) {
  #pragma unroll
  for (int rr = 0; rr < 8; ++rr) {
    const int row = row0 + rr + half * 8;
    float v = acc[rr];
    if (MODE == 0)      v += bias[col];
    else if (MODE == 2) v += bias[col] * aux[(size_t)row * N + col];
    Cout[(size_t)row * N + col] = v;
  }
}

// ---------------------------------------------------------------------------
// WMMA GEMM:  Cout(M,N) = X(M,K) * W(N,K)^T  [+ epilogue]
// One wave computes a 32x64 output strip (2 M-tiles x 4 N-tiles, 8 accums):
// per 32-K chunk: 2 A-frags + 4 B-frags feed 8 WMMAs (0.75 loads/WMMA).
//   MODE 0: + bias[n]
//   MODE 1: (none)
//   MODE 2: + bias[n] * aux[row*N + n]      (y = hs*C^T + D .* u_proj)
// All dims divide 32/64 for every call; no divergent branches, so EXEC is
// all-ones around every WMMA (ISA 7.12 requirement).
// ---------------------------------------------------------------------------
template <int MODE>
__global__ void wmma_gemm_kernel(const float* __restrict__ X,
                                 const float* __restrict__ W,
                                 const float* __restrict__ bias,
                                 const float* __restrict__ aux,
                                 float* __restrict__ Cout,
                                 int M, int N, int K) {
  const int warp = threadIdx.x >> 5;
  const int lane = threadIdx.x & 31;
  const int strips_n = N >> 6;                          // 64-wide strips
  const int strip = blockIdx.x * (blockDim.x >> 5) + warp;
  const int tm = strip / strips_n;                      // 32-row strip index
  const int tn = strip - tm * strips_n;

  const int half = lane >> 4;      // 0: lanes 0-15, 1: lanes 16-31
  const int r    = lane & 15;

  const float* xrow0 = X + (size_t)(tm * 32 + r) * K;        // A rows (M-tile 0)
  const float* xrow1 = xrow0 + (size_t)16 * K;               // A rows (M-tile 1)
  const float* wrow0 = W + (size_t)(tn * 64 + r) * K;        // B cols n0+r
  const float* wrow1 = wrow0 + (size_t)16 * K;
  const float* wrow2 = wrow0 + (size_t)32 * K;
  const float* wrow3 = wrow0 + (size_t)48 * K;

  v8f c00 = {}, c01 = {}, c02 = {}, c03 = {};
  v8f c10 = {}, c11 = {}, c12 = {}, c13 = {};
  for (int k0 = 0; k0 < K; k0 += 32) {
    v16bf a0 = load_a_frag(xrow0, k0, half);
    v16bf a1 = load_a_frag(xrow1, k0, half);
    v16bf b0 = load_b_frag(wrow0, k0, half);
    v16bf b1 = load_b_frag(wrow1, k0, half);
    v16bf b2 = load_b_frag(wrow2, k0, half);
    v16bf b3 = load_b_frag(wrow3, k0, half);
    c00 = __builtin_amdgcn_wmma_f32_16x16x32_bf16(false, a0, false, b0, (short)0, c00, false, false);
    c10 = __builtin_amdgcn_wmma_f32_16x16x32_bf16(false, a1, false, b0, (short)0, c10, false, false);
    c01 = __builtin_amdgcn_wmma_f32_16x16x32_bf16(false, a0, false, b1, (short)0, c01, false, false);
    c11 = __builtin_amdgcn_wmma_f32_16x16x32_bf16(false, a1, false, b1, (short)0, c11, false, false);
    c02 = __builtin_amdgcn_wmma_f32_16x16x32_bf16(false, a0, false, b2, (short)0, c02, false, false);
    c12 = __builtin_amdgcn_wmma_f32_16x16x32_bf16(false, a1, false, b2, (short)0, c12, false, false);
    c03 = __builtin_amdgcn_wmma_f32_16x16x32_bf16(false, a0, false, b3, (short)0, c03, false, false);
    c13 = __builtin_amdgcn_wmma_f32_16x16x32_bf16(false, a1, false, b3, (short)0, c13, false, false);
  }

  const int row0 = tm * 32;
  const int row1 = row0 + 16;
  const int col0 = tn * 64 + r;
  store_tile<MODE>(c00, Cout, bias, aux, row0, col0,      half, N);
  store_tile<MODE>(c01, Cout, bias, aux, row0, col0 + 16, half, N);
  store_tile<MODE>(c02, Cout, bias, aux, row0, col0 + 32, half, N);
  store_tile<MODE>(c03, Cout, bias, aux, row0, col0 + 48, half, N);
  store_tile<MODE>(c10, Cout, bias, aux, row1, col0,      half, N);
  store_tile<MODE>(c11, Cout, bias, aux, row1, col0 + 16, half, N);
  store_tile<MODE>(c12, Cout, bias, aux, row1, col0 + 32, half, N);
  store_tile<MODE>(c13, Cout, bias, aux, row1, col0 + 48, half, N);
}

// ---------------------------------------------------------------------------
// delta = softplus(delta_t * W_dt + b_dt), elementwise over (B*L, S=128)
// ---------------------------------------------------------------------------
__global__ void delta_kernel(const float* __restrict__ dt,
                             const float* __restrict__ Wdt,
                             const float* __restrict__ bdt,
                             float* __restrict__ delta, int total) {
  int idx = blockIdx.x * blockDim.x + threadIdx.x;
  if (idx >= total) return;
  int s  = idx & 127;
  int bl = idx >> 7;
  float x = dt[bl] * Wdt[s] + bdt[s];
  delta[idx] = fmaxf(x, 0.0f) + log1pf(__expf(-fabsf(x)));  // stable softplus
}

// ---------------------------------------------------------------------------
// Sequential scan: one block per batch. A (128x128, 64KB) staged in LDS
// PRE-SCALED by log2(e), so the inner loop is pure v_exp_f32 (exp2) + FMA:
//   h[i] <- sum_j exp2(A2[i,j]*d[i]) * h[j] + d[i]*Bu[i]
// h ping-pongs in LDS; 8 lanes per row, shuffle-reduced; delta/Bu pipelined
// one step ahead + global_prefetch_b8 eight steps ahead; 1 barrier per step.
// ---------------------------------------------------------------------------
__global__ void scan_kernel(const float* __restrict__ A,
                            const float* __restrict__ h0,
                            const float* __restrict__ delta,
                            const float* __restrict__ Bu,
                            float* __restrict__ hs, int L) {
  extern __shared__ float smem[];
  float* A_lds = smem;              // 128*128 floats, pre-scaled by log2(e)
  float* hbuf  = smem + 128 * 128;  // 2 * 128 floats (ping-pong)

  const int b   = blockIdx.x;
  const int tid = threadIdx.x;      // 1024 threads = 32 wave32
  const float LOG2E = 1.44269504088896340736f;

  // Stage A into LDS, folding the exp->exp2 conversion into the constant
  {
    const float4* g4 = (const float4*)A;
    float4*       l4 = (float4*)A_lds;
    for (int i = tid; i < (128 * 128) / 4; i += 1024) {
      float4 v = g4[i];
      v.x *= LOG2E; v.y *= LOG2E; v.z *= LOG2E; v.w *= LOG2E;
      l4[i] = v;
    }
  }
  if (tid < 128) hbuf[tid] = h0[b * 128 + tid];
  __syncthreads();

  const int i = tid >> 3;   // state row 0..127
  const int p = tid & 7;    // 8 lanes per row, 16 K each
  const float4* Arow4 = (const float4*)&A_lds[i * 128 + p * 16];
  const size_t chain = (size_t)b * L * 128 + i;   // &delta[b, k, i] stride 128

  int cur = 0;
  float d  = delta[chain];
  float bu = Bu[chain];
  for (int k = 0; k < L; ++k) {
    // software-pipelined loads for step k+1 (independent of h)
    const int kn = (k + 1 < L) ? (k + 1) : k;
    const float d_next  = delta[chain + (size_t)kn * 128];
    const float bu_next = Bu[chain + (size_t)kn * 128];
    if (k + 8 < L) {  // lowers to global_prefetch_b8
      __builtin_prefetch(&delta[chain + (size_t)(k + 8) * 128], 0, 3);
      __builtin_prefetch(&Bu[chain + (size_t)(k + 8) * 128], 0, 3);
    }

    const float* h = hbuf + cur * 128;
    const float4* h4 = (const float4*)(h + p * 16);
    float sum = 0.0f;
    #pragma unroll
    for (int c = 0; c < 4; ++c) {
      float4 av = Arow4[c];
      float4 hv = h4[c];
      sum += __builtin_amdgcn_exp2f(av.x * d) * hv.x;
      sum += __builtin_amdgcn_exp2f(av.y * d) * hv.y;
      sum += __builtin_amdgcn_exp2f(av.z * d) * hv.z;
      sum += __builtin_amdgcn_exp2f(av.w * d) * hv.w;
    }
    // reduce across the 8 lanes of this row (stays inside the wave32)
    sum += __shfl_xor(sum, 1, 32);
    sum += __shfl_xor(sum, 2, 32);
    sum += __shfl_xor(sum, 4, 32);

    if (p == 0) {
      float hn = sum + d * bu;
      hbuf[(cur ^ 1) * 128 + i] = hn;
      hs[chain + (size_t)k * 128] = hn;
    }
    d = d_next; bu = bu_next;
    cur ^= 1;
    __syncthreads();
  }
}

// ---------------------------------------------------------------------------
extern "C" void kernel_launch(void* const* d_in, const int* in_sizes, int n_in,
                              void* d_out, int out_size, void* d_ws, size_t ws_size,
                              hipStream_t stream) {
  const float* u    = (const float*)d_in[0];
  const float* dt   = (const float*)d_in[1];
  const float* h0   = (const float*)d_in[2];
  const float* W_in = (const float*)d_in[3];
  const float* b_in = (const float*)d_in[4];
  const float* A    = (const float*)d_in[5];
  const float* Bm   = (const float*)d_in[6];
  const float* Cm   = (const float*)d_in[7];
  const float* D    = (const float*)d_in[8];
  const float* W_dt = (const float*)d_in[9];
  const float* b_dt = (const float*)d_in[10];
  float* y = (float*)d_out;

  const int B = 16, L = 1024, I = 512, H = 1024, S = 128;
  const int M = B * L;  // 16384

  char* ws = (char*)d_ws;
  float* uproj = (float*)ws;                                      // M*H
  float* Bu    = (float*)(ws + (size_t)M * H * 4);                // M*S
  float* delta = (float*)(ws + (size_t)M * H * 4 + (size_t)M * S * 4);
  float* hs    = (float*)(ws + (size_t)M * H * 4 + 2 * (size_t)M * S * 4);

  // 1) u_proj = u * W_in^T + b_in     (M x H, K=I)
  {
    int strips = (M / 32) * (H / 64);               // 8192 waves
    wmma_gemm_kernel<0><<<strips / 8, 256, 0, stream>>>(u, W_in, b_in, nullptr,
                                                        uproj, M, H, I);
  }
  // 2) Bu = u_proj * Bm^T             (M x S, K=H)
  {
    int strips = (M / 32) * (S / 64);               // 1024 waves
    wmma_gemm_kernel<1><<<strips / 8, 256, 0, stream>>>(uproj, Bm, nullptr, nullptr,
                                                        Bu, M, S, H);
  }
  // 3) delta = softplus(dt * W_dt + b_dt)
  {
    int total = M * S;                              // 2M
    delta_kernel<<<total / 256, 256, 0, stream>>>(dt, W_dt, b_dt, delta, total);
  }
  // 4) sequential state-space scan (A staged in LDS: 64KB + 1KB ping-pong h)
  {
    size_t shmem = (size_t)(128 * 128 + 2 * 128) * sizeof(float);
    scan_kernel<<<B, 1024, shmem, stream>>>(A, h0, delta, Bu, hs, L);
  }
  // 5) y = hs * C^T + D .* u_proj     (M x H, K=S)
  {
    int strips = (M / 32) * (H / 64);
    wmma_gemm_kernel<2><<<strips / 8, 256, 0, stream>>>(hs, Cm, D, uproj,
                                                        y, M, H, S);
  }
}
